// MultiHeadAttention_54107997995198
// MI455X (gfx1250) — matrix operand order repaired and checked by reading, not compiled, not run on hardware
//
#include <hip/hip_runtime.h>

#define B_   2
#define S_   2048
#define H_   2048
#define NH_  32
#define NKV_ 8
#define D_   64

typedef __bf16 bf16_t;
typedef __attribute__((ext_vector_type(16))) __bf16 v16bf;
typedef __attribute__((ext_vector_type(8)))  float  v8f;

struct __align__(16) U4 { unsigned x, y, z, w; };

// CDNA5 async global->LDS path (ASYNCcnt), guarded so absence degrades cleanly.
#if defined(__AMDGCN__) && __has_builtin(__builtin_amdgcn_global_load_async_to_lds_b128)
#define ASYNC_LDS 1
typedef int v4i_gcc __attribute__((vector_size(16)));
typedef __attribute__((address_space(1))) v4i_gcc as1_v4i;
typedef __attribute__((address_space(3))) v4i_gcc as3_v4i;
__device__ __forceinline__ void async_copy16(const void* g, void* l) {
  __builtin_amdgcn_global_load_async_to_lds_b128((as1_v4i*)g, (as3_v4i*)l, 0, 0);
}
__device__ __forceinline__ void wait_async0() {
#if __has_builtin(__builtin_amdgcn_s_wait_asynccnt)
  __builtin_amdgcn_s_wait_asynccnt(0);
#else
  asm volatile("s_wait_asynccnt 0x0" ::: "memory");
#endif
}
#endif

__device__ __forceinline__ bf16_t f2bf(float f) {
  union { float f; unsigned u; } v; v.f = f;
  unsigned r = (v.u + 0x7FFFu + ((v.u >> 16) & 1u)) >> 16;
  union { unsigned short s; bf16_t b; } o; o.s = (unsigned short)r;
  return o.b;
}
__device__ __forceinline__ float bf2f(bf16_t b) {
  union { bf16_t b; unsigned short s; } i; i.b = b;
  union { unsigned u; float f; } o; o.u = ((unsigned)i.s) << 16;
  return o.f;
}

// ---------------- fp32 -> bf16 bulk convert ----------------
__global__ void to_bf16_kernel(const float* __restrict__ in,
                               bf16_t* __restrict__ out, int n) {
  int i = blockIdx.x * blockDim.x + threadIdx.x;
  int stride = gridDim.x * blockDim.x;
  for (; i < n; i += stride) out[i] = f2bf(in[i]);
}

// ------------- weight convert + transpose: WT[n][k] = W[k][map(n)] -------------
__global__ void make_wt_kernel(const float* __restrict__ W, int ld,
                               bf16_t* __restrict__ WT, int K, int mode) {
  int k = blockIdx.x * blockDim.x + threadIdx.x;
  int n = blockIdx.y;
  if (k >= K) return;
  int col = (mode == 0) ? n : ((n >> 6) * 128 + ((mode == 2) ? 64 : 0) + (n & 63));
  WT[(size_t)n * K + k] = f2bf(W[(size_t)k * ld + col]);
}

// ---------------- bf16 WMMA GEMM: C[M,N] = A[M,K] * BT[N,K]^T ----------------
// 128x128 block tile, 8 waves; wave computes 32x64 (2 M-frags x 4 N-tiles).
// Async path: double-buffered LDS fed by GLOBAL_LOAD_ASYNC_TO_LDS_B128.
__global__ __launch_bounds__(256) void gemm_bf16_wmma(
    const bf16_t* __restrict__ A, const bf16_t* __restrict__ BT,
    void* __restrict__ C, int M, int N, int K, int c_f32) {
#ifdef ASYNC_LDS
  __shared__ bf16_t lA[2][128][40];
  __shared__ bf16_t lB[2][128][40];
#else
  __shared__ bf16_t lA[1][128][40];
  __shared__ bf16_t lB[1][128][40];
#endif

  const int tid  = threadIdx.x;
  const int wave = tid >> 5;
  const int lane = tid & 31;
  const int h    = lane >> 4;
  const int lm   = lane & 15;
  const int wm   = wave & 3;   // 32-row slice
  const int wn   = wave >> 2;  // 64-col slice
  const int m0   = blockIdx.y * 128;
  const int n0   = blockIdx.x * 128;

  v8f acc[2][4] = {};

  const int sr = tid >> 1;          // staging row (2 threads/row, 128 rows)
  const int sc = (tid & 1) * 16;    // staging col chunk
  const bf16_t* gA = A  + (size_t)(m0 + sr) * K + sc;
  const bf16_t* gB = BT + (size_t)(n0 + sr) * K + sc;

#ifdef ASYNC_LDS
  // prologue: fill buffer 0
  async_copy16(gA,     &lA[0][sr][sc]);
  async_copy16(gA + 8, &lA[0][sr][sc + 8]);
  async_copy16(gB,     &lB[0][sr][sc]);
  async_copy16(gB + 8, &lB[0][sr][sc + 8]);
#endif

  int p = 0;
  for (int k0 = 0; k0 < K; k0 += 32) {
#ifdef ASYNC_LDS
    wait_async0();
    __syncthreads();                       // buffer p ready; prev compute done
    if (k0 + 32 < K) {
      const bf16_t* nA = gA + k0 + 32;
      const bf16_t* nB = gB + k0 + 32;
      async_copy16(nA,     &lA[p ^ 1][sr][sc]);
      async_copy16(nA + 8, &lA[p ^ 1][sr][sc + 8]);
      async_copy16(nB,     &lB[p ^ 1][sr][sc]);
      async_copy16(nB + 8, &lB[p ^ 1][sr][sc + 8]);
      __builtin_prefetch((const char*)nA + 64, 0, 3);
      __builtin_prefetch((const char*)nB + 64, 0, 3);
    }
#else
    const U4* sa = (const U4*)(gA + k0);
    U4 a0 = sa[0], a1 = sa[1];
    const U4* sb = (const U4*)(gB + k0);
    U4 b0 = sb[0], b1 = sb[1];
    __builtin_prefetch((const char*)sa + 64, 0, 3);
    __builtin_prefetch((const char*)sb + 64, 0, 3);
    __syncthreads();
    *(U4*)&lA[0][sr][sc] = a0;  *(U4*)&lA[0][sr][sc + 8] = a1;
    *(U4*)&lB[0][sr][sc] = b0;  *(U4*)&lB[0][sr][sc + 8] = b1;
    __syncthreads();
#endif

    v16bf av[2];
#pragma unroll
    for (int mi = 0; mi < 2; ++mi) {
      const bf16_t* pa = &lA[p][wm * 32 + mi * 16 + lm][h * 8];
      ((U4*)&av[mi])[0] = *(const U4*)pa;          // K = h*8 ..
      ((U4*)&av[mi])[1] = *(const U4*)(pa + 16);   // K = 16+h*8 ..
    }
#pragma unroll
    for (int nt = 0; nt < 4; ++nt) {
      v16bf bv;
      const bf16_t* pb = &lB[p][wn * 64 + nt * 16 + lm][h * 8];
      ((U4*)&bv)[0] = *(const U4*)pb;
      ((U4*)&bv)[1] = *(const U4*)(pb + 16);
#pragma unroll
      for (int mi = 0; mi < 2; ++mi)
        acc[mi][nt] = __builtin_amdgcn_wmma_f32_16x16x32_bf16(
            false, av[mi], false, bv, (short)0, acc[mi][nt], false, false);
    }
#ifdef ASYNC_LDS
    p ^= 1;
#endif
  }

#pragma unroll
  for (int mi = 0; mi < 2; ++mi)
#pragma unroll
    for (int nt = 0; nt < 4; ++nt)
#pragma unroll
      for (int r = 0; r < 8; ++r) {
        int row = m0 + wm * 32 + mi * 16 + h * 8 + r;
        int col = n0 + wn * 64 + nt * 16 + lm;
        float v = acc[mi][nt][r];
        if (c_f32) ((float*)C)[(size_t)row * N + col] = v;
        else       ((bf16_t*)C)[(size_t)row * N + col] = f2bf(v);
      }
}

// ---------------- RoPE (in-place on bf16 Q and K) ----------------
__global__ void rope_kernel(bf16_t* __restrict__ Q, bf16_t* __restrict__ Kb,
                            const int* __restrict__ pos, int total) {
  int idx = blockIdx.x * blockDim.x + threadIdx.x;
  if (idx >= total) return;
  int i  = idx & 31;
  int t  = idx >> 5;
  int hd = t % (NH_ + NKV_);
  int bs = t / (NH_ + NKV_);
  float p   = (float)pos[bs];
  float inv = __powf(10000.0f, -((float)(2 * i)) / 64.0f);
  float ang = p * inv;
  float sn = __sinf(ang), cs = __cosf(ang);
  bf16_t* base; size_t off;
  if (hd < NH_) { base = Q;  off = (size_t)bs * (NH_  * D_) + (size_t)hd         * D_ + i; }
  else          { base = Kb; off = (size_t)bs * (NKV_ * D_) + (size_t)(hd - NH_) * D_ + i; }
  float x1 = bf2f(base[off]);
  float x2 = bf2f(base[off + 32]);
  base[off]      = f2bf(x1 * cs - x2 * sn);
  base[off + 32] = f2bf(x2 * cs + x1 * sn);
}

// ---------------- causal flash attention, bf16 WMMA, f32 online softmax ----------------
__global__ __launch_bounds__(128) void flash_attn_kernel(
    const bf16_t* __restrict__ Q, const bf16_t* __restrict__ Kb,
    const bf16_t* __restrict__ Vb, bf16_t* __restrict__ Ctx) {
  __shared__ bf16_t lQ[64][72];
  __shared__ bf16_t lK[64][72];
  __shared__ bf16_t lVt[64][72];       // transposed: [d][key]
  __shared__ bf16_t lP[4][16][72];     // per-wave probs relayout buffer

  const int tid  = threadIdx.x;
  const int wave = tid >> 5;
  const int lane = tid & 31;
  const int h    = lane >> 4;
  const int lm   = lane & 15;
  const int qt   = blockIdx.x;
  const int bh   = blockIdx.y;
  const int b    = bh >> 5;
  const int hq   = bh & 31;
  const int hkv  = hq >> 2;

  const int sr  = tid >> 1;
  const int scb = (tid & 1) * 32;

  {
    const U4* src = (const U4*)(Q + (size_t)(b * S_ + qt * 64 + sr) * (NH_ * D_) + hq * D_ + scb);
#pragma unroll
    for (int j = 0; j < 4; ++j) *(U4*)&lQ[sr][scb + j * 8] = src[j];
  }
  __syncthreads();

  v16bf aQ[2];
#pragma unroll
  for (int ks = 0; ks < 2; ++ks) {
    const bf16_t* p = &lQ[wave * 16 + lm][ks * 32 + h * 8];
    ((U4*)&aQ[ks])[0] = *(const U4*)p;
    ((U4*)&aQ[ks])[1] = *(const U4*)(p + 16);
  }

  float mi[8], li[8];
  v8f acc[4] = {};
#pragma unroll
  for (int r = 0; r < 8; ++r) { mi[r] = -3.0e38f; li[r] = 0.0f; }

  for (int kt = 0; kt <= qt; ++kt) {
    const U4* srcK = (const U4*)(Kb + (size_t)(b * S_ + kt * 64 + sr) * (NKV_ * D_) + hkv * D_ + scb);
    const U4* srcV = (const U4*)(Vb + (size_t)(b * S_ + kt * 64 + sr) * (NKV_ * D_) + hkv * D_ + scb);
    if (kt < qt) {  // prefetch next K/V tiles
      __builtin_prefetch((const char*)srcK + (size_t)64 * NKV_ * D_ * 2, 0, 3);
      __builtin_prefetch((const char*)srcV + (size_t)64 * NKV_ * D_ * 2, 0, 3);
    }
    U4 vdr[4];
#pragma unroll
    for (int j = 0; j < 4; ++j) vdr[j] = srcV[j];
#ifndef ASYNC_LDS
    U4 kd[4];
#pragma unroll
    for (int j = 0; j < 4; ++j) kd[j] = srcK[j];
#endif

    __syncthreads();   // previous tile's compute finished
#ifdef ASYNC_LDS
#pragma unroll
    for (int j = 0; j < 4; ++j)
      async_copy16(srcK + j, &lK[sr][scb + j * 8]);
#else
#pragma unroll
    for (int j = 0; j < 4; ++j) *(U4*)&lK[sr][scb + j * 8] = kd[j];
#endif
    {
      union { U4 u[4]; bf16_t e[32]; } vv;
#pragma unroll
      for (int j = 0; j < 4; ++j) vv.u[j] = vdr[j];
#pragma unroll
      for (int j = 0; j < 32; ++j) lVt[scb + j][sr] = vv.e[j];   // transpose store
    }
#ifdef ASYNC_LDS
    wait_async0();
#endif
    __syncthreads();

    // S = Q * K^T
    v8f s[4];
#pragma unroll
    for (int nt = 0; nt < 4; ++nt) {
      v16bf bk0, bk1;
      const bf16_t* p0 = &lK[nt * 16 + lm][h * 8];
      ((U4*)&bk0)[0] = *(const U4*)p0;
      ((U4*)&bk0)[1] = *(const U4*)(p0 + 16);
      const bf16_t* p1 = &lK[nt * 16 + lm][32 + h * 8];
      ((U4*)&bk1)[0] = *(const U4*)p1;
      ((U4*)&bk1)[1] = *(const U4*)(p1 + 16);
      v8f z = {};
      v8f t0 = __builtin_amdgcn_wmma_f32_16x16x32_bf16(
          false, aQ[0], false, bk0, (short)0, z, false, false);
      s[nt] = __builtin_amdgcn_wmma_f32_16x16x32_bf16(
          false, aQ[1], false, bk1, (short)0, t0, false, false);
    }

    float sv[4][8];
#pragma unroll
    for (int nt = 0; nt < 4; ++nt)
#pragma unroll
      for (int r = 0; r < 8; ++r) sv[nt][r] = s[nt][r] * 0.125f;

    if (kt == qt) {
#pragma unroll
      for (int nt = 0; nt < 4; ++nt)
#pragma unroll
        for (int r = 0; r < 8; ++r)
          if (nt * 16 + lm > wave * 16 + h * 8 + r) sv[nt][r] = -3.0e38f;
    }

    float rm[8];
#pragma unroll
    for (int r = 0; r < 8; ++r)
      rm[r] = fmaxf(fmaxf(sv[0][r], sv[1][r]), fmaxf(sv[2][r], sv[3][r]));
#pragma unroll
    for (int m = 1; m < 16; m <<= 1)
#pragma unroll
      for (int r = 0; r < 8; ++r) rm[r] = fmaxf(rm[r], __shfl_xor(rm[r], m, 32));

    float nm[8], so[8], rs[8];
#pragma unroll
    for (int r = 0; r < 8; ++r) {
      nm[r] = fmaxf(mi[r], rm[r]);
      so[r] = __expf(mi[r] - nm[r]);
      rs[r] = 0.0f;
    }
#pragma unroll
    for (int nt = 0; nt < 4; ++nt)
#pragma unroll
      for (int r = 0; r < 8; ++r) {
        float p = __expf(sv[nt][r] - nm[r]);
        rs[r] += p;
        lP[wave][h * 8 + r][nt * 16 + lm] = f2bf(p);
      }
#pragma unroll
    for (int m = 1; m < 16; m <<= 1)
#pragma unroll
      for (int r = 0; r < 8; ++r) rs[r] += __shfl_xor(rs[r], m, 32);
#pragma unroll
    for (int r = 0; r < 8; ++r) { li[r] = li[r] * so[r] + rs[r]; mi[r] = nm[r]; }
#pragma unroll
    for (int nt = 0; nt < 4; ++nt)
#pragma unroll
      for (int r = 0; r < 8; ++r) acc[nt][r] *= so[r];

    __syncthreads();   // P relayout visible

    v16bf aP[2];
#pragma unroll
    for (int ks = 0; ks < 2; ++ks) {
      const bf16_t* p = &lP[wave][lm][ks * 32 + h * 8];
      ((U4*)&aP[ks])[0] = *(const U4*)p;
      ((U4*)&aP[ks])[1] = *(const U4*)(p + 16);
    }
#pragma unroll
    for (int nt = 0; nt < 4; ++nt) {
      v16bf bv0, bv1;
      const bf16_t* p0 = &lVt[nt * 16 + lm][h * 8];
      ((U4*)&bv0)[0] = *(const U4*)p0;
      ((U4*)&bv0)[1] = *(const U4*)(p0 + 16);
      const bf16_t* p1 = &lVt[nt * 16 + lm][32 + h * 8];
      ((U4*)&bv1)[0] = *(const U4*)p1;
      ((U4*)&bv1)[1] = *(const U4*)(p1 + 16);
      acc[nt] = __builtin_amdgcn_wmma_f32_16x16x32_bf16(
          false, aP[0], false, bv0, (short)0, acc[nt], false, false);
      acc[nt] = __builtin_amdgcn_wmma_f32_16x16x32_bf16(
          false, aP[1], false, bv1, (short)0, acc[nt], false, false);
    }
  }

#pragma unroll
  for (int nt = 0; nt < 4; ++nt)
#pragma unroll
    for (int r = 0; r < 8; ++r) {
      int row = qt * 64 + wave * 16 + h * 8 + r;
      int col = hq * D_ + nt * 16 + lm;
      float v = acc[nt][r] / li[r];
      Ctx[(size_t)(b * S_ + row) * (NH_ * D_) + col] = f2bf(v);
    }
}

// ---------------- host-side orchestration ----------------
extern "C" void kernel_launch(void* const* d_in, const int* in_sizes, int n_in,
                              void* d_out, int out_size, void* d_ws, size_t ws_size,
                              hipStream_t stream) {
  (void)in_sizes; (void)n_in; (void)out_size; (void)ws_size;
  const float* hidden = (const float*)d_in[0];
  const int*   pos    = (const int*)d_in[1];
  const float* Wq     = (const float*)d_in[2];
  const float* Wkv    = (const float*)d_in[3];
  const float* Wo     = (const float*)d_in[4];
  float* out = (float*)d_out;

  char* ws = (char*)d_ws;
  size_t off = 0;
  auto alloc = [&](size_t bytes) { void* p = ws + off; off += bytes; return p; };
  bf16_t* Xbf  = (bf16_t*)alloc((size_t)4096 * 2048 * 2);
  bf16_t* WqT  = (bf16_t*)alloc((size_t)2048 * 2048 * 2);
  bf16_t* WkT  = (bf16_t*)alloc((size_t)512  * 2048 * 2);
  bf16_t* WvT  = (bf16_t*)alloc((size_t)512  * 2048 * 2);
  bf16_t* WoT  = (bf16_t*)alloc((size_t)2048 * 2048 * 2);
  bf16_t* Qb   = (bf16_t*)alloc((size_t)4096 * 2048 * 2);
  bf16_t* Kbuf = (bf16_t*)alloc((size_t)4096 * 512  * 2);
  bf16_t* Vbuf = (bf16_t*)alloc((size_t)4096 * 512  * 2);
  bf16_t* Cx   = (bf16_t*)alloc((size_t)4096 * 2048 * 2);

  to_bf16_kernel<<<4096, 256, 0, stream>>>(hidden, Xbf, 4096 * 2048);
  make_wt_kernel<<<dim3(8, 2048), 256, 0, stream>>>(Wq,  2048, WqT, 2048, 0);
  make_wt_kernel<<<dim3(8, 512),  256, 0, stream>>>(Wkv, 1024, WkT, 2048, 1);
  make_wt_kernel<<<dim3(8, 512),  256, 0, stream>>>(Wkv, 1024, WvT, 2048, 2);
  make_wt_kernel<<<dim3(8, 2048), 256, 0, stream>>>(Wo,  2048, WoT, 2048, 0);

  gemm_bf16_wmma<<<dim3(16, 32), 256, 0, stream>>>(Xbf, WqT, Qb,   4096, 2048, 2048, 0);
  gemm_bf16_wmma<<<dim3(4, 32),  256, 0, stream>>>(Xbf, WkT, Kbuf, 4096, 512,  2048, 0);
  gemm_bf16_wmma<<<dim3(4, 32),  256, 0, stream>>>(Xbf, WvT, Vbuf, 4096, 512,  2048, 0);

  {
    int total = B_ * S_ * (NH_ + NKV_) * (D_ / 2);
    rope_kernel<<<(total + 255) / 256, 256, 0, stream>>>(Qb, Kbuf, pos, total);
  }

  flash_attn_kernel<<<dim3(S_ / 64, B_ * NH_), 128, 0, stream>>>(Qb, Kbuf, Vbuf, Cx);

  gemm_bf16_wmma<<<dim3(16, 32), 256, 0, stream>>>(Cx, WoT, out, 4096, 2048, 2048, 1);
}
